// RelPosMultiheadAttention_77060303224893
// MI455X (gfx1250) — compile-verified, matrix operand order checked by reading, not run on hardware
//
#include <hip/hip_runtime.h>

#define D_MODEL 768
#define N_HEADS 12
#define HEAD_DIM 64
#define SEQ_T 1024
#define BATCH 4
#define NTOK (BATCH * SEQ_T) /* 4096 */
#define N_REL 1024

typedef __attribute__((ext_vector_type(16))) _Float16 v16h;
typedef __attribute__((ext_vector_type(8)))  float    v8f;
typedef __attribute__((ext_vector_type(4)))  unsigned v4u;
typedef __attribute__((ext_vector_type(8)))  int      v8i;
typedef __attribute__((ext_vector_type(4)))  int      v4i;

// ---------------------------------------------------------------------------
// CDNA5 16-bit A/B fragment loader (CDNA5 ISA §7.12.2).
// Logical matrix is 16 rows x 32 K, row-major with leading dim `ld` (f16 units).
// Lane layout: row = lane%16 ; K-offset 0 (lanes 0-15) or 8 (lanes 16-31);
// VGPR 0-3 hold K pairs {koff..koff+7}, VGPR 4-7 hold {16+koff..16+koff+7}.
// Each half is 8 contiguous f16 = 16B -> lowers to two global/ds b128 loads.
// ---------------------------------------------------------------------------
__device__ __forceinline__ v16h load_frag(const _Float16* p, int ld, int lane) {
  const int r = lane & 15;
  const int koff = (lane & 16) ? 8 : 0;
  const _Float16* row = p + (size_t)r * ld;
  v16h f;
#pragma unroll
  for (int i = 0; i < 4; ++i) {
    f[2 * i]     = row[koff + 2 * i];
    f[2 * i + 1] = row[koff + 2 * i + 1];
  }
#pragma unroll
  for (int i = 0; i < 4; ++i) {
    f[8 + 2 * i] = row[16 + koff + 2 * i];
    f[9 + 2 * i] = row[16 + koff + 2 * i + 1];
  }
  return f;
}

__device__ __forceinline__ v8f wmma_acc(v16h a, v16h b, v8f c) {
  // D = A(16x32 f16) * B(32x16 f16) + C(16x16 f32)
  return __builtin_amdgcn_wmma_f32_16x16x32_f16(
      /*neg_a=*/false, a, /*neg_b=*/false, b,
      /*c_mod=*/(short)0, c, /*reuse_a=*/false, /*reuse_b=*/false);
}

// ---------------------------------------------------------------------------
// Tensor Data Mover: 2D tile (tile_w elems x tile_h rows, 2-byte elements)
// from global row-major (row stride = stride_elems) into LDS, packed rows.
// D# layout per CDNA5 ISA:
//  group0: [1:0]=count=1 | [63:32]=lds_addr | [120:64]=global_addr | [127:126]=2
//  group1: [17:16]=data_size(1=2B) | [79:48]=tensor_dim0 | [111:80]=tensor_dim1
//          [127:112]=tile_dim0 | [143:128]=tile_dim1 | [207:160]=dim0_stride
//  groups 2/3 (and trailing group): zero (2D tensor).
// Tracked with TENSORcnt; EXEC ignored (issue from one wave only).
// ---------------------------------------------------------------------------
__device__ __forceinline__ void tdm_load_2d(unsigned lds_addr, const void* gptr,
                                            unsigned tile_w, unsigned tile_h,
                                            unsigned stride_elems) {
  const unsigned long long ga = (unsigned long long)(size_t)gptr;
  v4u g0;
  g0[0] = 1u;                                        // count=1, user mode
  g0[1] = lds_addr;                                  // LDS byte address
  g0[2] = (unsigned)(ga & 0xffffffffu);              // global_addr[31:0]
  g0[3] = (unsigned)((ga >> 32) & 0x01ffffffu) | (2u << 30);  // [56:32] | type=2
  v8i g1;
  g1[0] = (int)(1u << 16);                           // data_size = 1 (2 bytes)
  g1[1] = (int)((tile_w & 0xffffu) << 16);           // tensor_dim0 lo16
  g1[2] = (int)(((tile_w >> 16) & 0xffffu) |         // tensor_dim0 hi16
                ((tile_h & 0xffffu) << 16));         // tensor_dim1 lo16
  g1[3] = (int)(((tile_h >> 16) & 0xffffu) |         // tensor_dim1 hi16
                ((tile_w & 0xffffu) << 16));         // tile_dim0
  g1[4] = (int)(tile_h & 0xffffu);                   // tile_dim1 (tile_dim2=0)
  g1[5] = (int)stride_elems;                         // tensor_dim0_stride lo32
  g1[6] = 0;                                         // stride hi / dim1_stride
  g1[7] = 0;
  const v4i z4 = {0, 0, 0, 0};
  const v8i z8 = {0, 0, 0, 0, 0, 0, 0, 0};
  __builtin_amdgcn_tensor_load_to_lds(g0, g1, z4, z4, z8, 0);
}

// ---------------------------------------------------------------------------
// f32 -> f16 elementwise convert
// ---------------------------------------------------------------------------
__global__ __launch_bounds__(256) void cvt_f32_to_f16(
    const float* __restrict__ src, _Float16* __restrict__ dst, int n) {
  int i = blockIdx.x * blockDim.x + threadIdx.x;
  if (i < n) dst[i] = (_Float16)src[i];
}

// ---------------------------------------------------------------------------
// Fused QKV projection:  q/k/v = x @ W^T + b  (M=4096, N=768, K=768)
// One wave -> one 16(M) x 64(N) strip of one of the three outputs.
// ---------------------------------------------------------------------------
__global__ __launch_bounds__(256) void qkv_proj_kernel(
    const _Float16* __restrict__ xh,
    const _Float16* __restrict__ wqh, const _Float16* __restrict__ wkh,
    const _Float16* __restrict__ wvh,
    const float* __restrict__ bq, const float* __restrict__ bk,
    const float* __restrict__ bv,
    _Float16* __restrict__ qh, _Float16* __restrict__ kh,
    _Float16* __restrict__ vT) {
  const int lane = threadIdx.x & 31;
  const int wid = blockIdx.x * (blockDim.x >> 5) + (threadIdx.x >> 5);
  const int mat = wid / (256 * 12);       // 0=Q 1=K 2=V
  const int rem = wid % (256 * 12);
  const int mtile = rem / 12;             // 16-row tile of 4096 tokens
  const int nstrip = rem % 12;            // 64-col strip of 768 features

  const _Float16* W = (mat == 0) ? wqh : (mat == 1) ? wkh : wvh;
  const float* bias = (mat == 0) ? bq : (mat == 1) ? bk : bv;

  v8f zero8 = {0.f, 0.f, 0.f, 0.f, 0.f, 0.f, 0.f, 0.f};
  v8f acc[4] = {zero8, zero8, zero8, zero8};

  const _Float16* abase = xh + (size_t)(mtile * 16) * D_MODEL;
  for (int kk = 0; kk < D_MODEL; kk += 64) {
    if (kk + 64 < D_MODEL) {
      __builtin_prefetch(abase + kk + 64, 0, 3);  // near-cache prefetch
    }
    v16h a0 = load_frag(abase + kk, D_MODEL, lane);
    v16h a1 = load_frag(abase + kk + 32, D_MODEL, lane);
    v16h b0[4], b1[4];
#pragma unroll
    for (int nt = 0; nt < 4; ++nt) {
      const _Float16* bb = W + (size_t)(nstrip * 64 + nt * 16) * D_MODEL + kk;
      b0[nt] = load_frag(bb, D_MODEL, lane);
      b1[nt] = load_frag(bb + 32, D_MODEL, lane);
    }
#pragma unroll
    for (int nt = 0; nt < 4; ++nt) acc[nt] = wmma_acc(a0, b0[nt], acc[nt]);
#pragma unroll
    for (int nt = 0; nt < 4; ++nt) acc[nt] = wmma_acc(a1, b1[nt], acc[nt]);
  }

  // D-fragment: VGPR r holds row r (lanes 0-15) / row r+8 (lanes 16-31),
  // column = lane%16 within the 16-wide tile.
  const int r_add = (lane & 16) ? 8 : 0;
#pragma unroll
  for (int nt = 0; nt < 4; ++nt) {
    const int col = nstrip * 64 + nt * 16 + (lane & 15);
    const float bval = bias[col];
    const int hh = col >> 6, dd = col & 63;
#pragma unroll
    for (int r = 0; r < 8; ++r) {
      const int tok = mtile * 16 + r + r_add;
      const int bb = tok >> 10, tt = tok & 1023;
      const float val = acc[nt][r] + bval;
      if (mat == 0) {
        qh[((size_t)(bb * N_HEADS + hh) * SEQ_T + tt) * HEAD_DIM + dd] =
            (_Float16)(val * 0.125f);  // fold 1/sqrt(64)
      } else if (mat == 1) {
        kh[((size_t)(bb * N_HEADS + hh) * SEQ_T + tt) * HEAD_DIM + dd] =
            (_Float16)val;
      } else {
        vT[((size_t)(bb * N_HEADS + hh) * HEAD_DIM + dd) * SEQ_T + tt] =
            (_Float16)val;
      }
    }
  }
}

// ---------------------------------------------------------------------------
// Flash attention with relative-position bias, TDM-staged K/V.
// Grid: 48 (b,h) x 8 q-blocks of 128 rows. 8 waves/block, 16 q-rows each.
// All 8 waves share each 32-token K/V tile, so wave 0 DMAs the tiles into
// double-buffered LDS via tensor_load_to_lds (TENSORcnt), and every wave
// reads its WMMA fragments from LDS (ds_load_b128). rel_emb lives in LDS as
// f16 (bias magnitudes ~0.02; f16 error negligible).
// LDS: 24K (rel) + 8K (P stage) + 8K (K dbuf) + 8K (V dbuf) = 48 KB.
// ---------------------------------------------------------------------------
__global__ __launch_bounds__(256) void attn_kernel(
    const _Float16* __restrict__ qh, const _Float16* __restrict__ kh,
    const _Float16* __restrict__ vT, const int* __restrict__ rel_ids,
    const unsigned char* __restrict__ kpm, const float* __restrict__ rel_emb,
    _Float16* __restrict__ oh) {
  __shared__ _Float16 srel[N_REL * N_HEADS];              // 24 KB
  __shared__ __align__(16) _Float16 pstage[8][16 * 32];   // 8 KB
  __shared__ __align__(16) _Float16 kbuf[2][32 * HEAD_DIM];  // 8 KB
  __shared__ __align__(16) _Float16 vbuf[2][HEAD_DIM * 32];  // 8 KB

  const int tid = threadIdx.x;
  for (int i = tid; i < N_REL * N_HEADS; i += 256)
    srel[i] = (_Float16)rel_emb[i];

  const int lane = tid & 31;
  const int w = tid >> 5;
  const int bh = blockIdx.x >> 3;  // 0..47
  const int qt = blockIdx.x & 7;
  const int b = bh / N_HEADS, h = bh % N_HEADS;
  const int qbase = qt * 128 + w * 16;
  const int r_add = (lane & 16) ? 8 : 0;

  const _Float16* kbase = kh + (size_t)bh * SEQ_T * HEAD_DIM;
  const _Float16* vbase = vT + (size_t)bh * HEAD_DIM * SEQ_T;

  // Prime buffer 0 with k-tile 0 (wave 0 only; TDM ignores EXEC).
  if (w == 0) {
    // K tile: 32 rows x 64 f16, row stride 64  -> packed 32x64 in LDS.
    tdm_load_2d((unsigned)(size_t)&kbuf[0][0], kbase, HEAD_DIM, 32, HEAD_DIM);
    // V tile: 64 rows (dims) x 32 f16, row stride 1024 -> packed 64x32 in LDS.
    tdm_load_2d((unsigned)(size_t)&vbuf[0][0], vbase, 32, HEAD_DIM, SEQ_T);
    __builtin_amdgcn_s_wait_tensorcnt(0);
  }
  __syncthreads();  // rel table + first K/V tiles ready

  const _Float16* qptr = qh + ((size_t)bh * SEQ_T + qbase) * HEAD_DIM;
  const v16h qa0 = load_frag(qptr, HEAD_DIM, lane);
  const v16h qa1 = load_frag(qptr + 32, HEAD_DIM, lane);

  float rowmax[8], rowsum[8];
  v8f zero8 = {0.f, 0.f, 0.f, 0.f, 0.f, 0.f, 0.f, 0.f};
  v8f acc[4] = {zero8, zero8, zero8, zero8};
#pragma unroll
  for (int r = 0; r < 8; ++r) { rowmax[r] = -1e30f; rowsum[r] = 0.f; }

  const int* ridb = rel_ids + (size_t)b * SEQ_T * SEQ_T;
  const unsigned char* kpb = kpm + (size_t)b * SEQ_T;
  _Float16* P = pstage[w];

  for (int jt = 0; jt < SEQ_T / 32; ++jt) {
    const int kb0 = jt * 32;
    const int bi = jt & 1;

    // Kick DMA for the next tile into the other buffer.
    const bool have_next = (jt + 1) < (SEQ_T / 32);
    if (w == 0 && have_next) {
      const int nb0 = kb0 + 32;
      tdm_load_2d((unsigned)(size_t)&kbuf[bi ^ 1][0],
                  kbase + (size_t)nb0 * HEAD_DIM, HEAD_DIM, 32, HEAD_DIM);
      tdm_load_2d((unsigned)(size_t)&vbuf[bi ^ 1][0],
                  vbase + nb0, 32, HEAD_DIM, SEQ_T);
    }

    // Independent global loads (rel ids + mask) issued early.
    const int col0 = kb0 + (lane & 15);
    const int col1 = col0 + 16;
    const bool m0 = kpb[col0] != 0;
    const bool m1 = kpb[col1] != 0;
    int rid0[8], rid1[8];
#pragma unroll
    for (int r = 0; r < 8; ++r) {
      const int qrow = qbase + r + r_add;
      rid0[r] = ridb[(size_t)qrow * SEQ_T + col0];
      rid1[r] = ridb[(size_t)qrow * SEQ_T + col1];
    }

    // K B-fragments from LDS (row = key token, ld = 64).
    const _Float16* kt = &kbuf[bi][0];
    v16h kf00 = load_frag(kt, HEAD_DIM, lane);
    v16h kf01 = load_frag(kt + 32, HEAD_DIM, lane);
    v16h kf10 = load_frag(kt + 16 * HEAD_DIM, HEAD_DIM, lane);
    v16h kf11 = load_frag(kt + 16 * HEAD_DIM + 32, HEAD_DIM, lane);

    // V B-fragments from LDS: row = hd dim, ld = 32, contiguous k-pairs.
    const _Float16* vt = &vbuf[bi][0];
    v16h vf[4];
#pragma unroll
    for (int n = 0; n < 4; ++n) vf[n] = load_frag(vt + n * 16 * 32, 32, lane);

    // Score WMMAs.
    v8f c0 = wmma_acc(qa0, kf00, zero8);
    c0 = wmma_acc(qa1, kf01, c0);
    v8f c1 = wmma_acc(qa0, kf10, zero8);
    c1 = wmma_acc(qa1, kf11, c1);

    float s0[8], s1[8];
#pragma unroll
    for (int r = 0; r < 8; ++r) {
      float t0 = c0[r] + (float)srel[rid0[r] * N_HEADS + h];
      float t1 = c1[r] + (float)srel[rid1[r] * N_HEADS + h];
      s0[r] = m0 ? -1e30f : t0;
      s1[r] = m1 ? -1e30f : t1;
    }

    // Online softmax. Row values live in one 16-lane half; xor masks 1,2,4,8
    // reduce within each half independently.
#pragma unroll
    for (int r = 0; r < 8; ++r) {
      float tm = fmaxf(s0[r], s1[r]);
#pragma unroll
      for (int m = 1; m < 16; m <<= 1) tm = fmaxf(tm, __shfl_xor(tm, m, 32));
      const float nm = fmaxf(rowmax[r], tm);
      const float p0 = __expf(s0[r] - nm);
      const float p1 = __expf(s1[r] - nm);
      float ps = p0 + p1;
#pragma unroll
      for (int m = 1; m < 16; m <<= 1) ps += __shfl_xor(ps, m, 32);
      const float corr = __expf(rowmax[r] - nm);
      rowsum[r] = rowsum[r] * corr + ps;
      rowmax[r] = nm;
#pragma unroll
      for (int n = 0; n < 4; ++n) acc[n][r] *= corr;
      // D-layout store of P tile (16x32) into per-wave LDS staging.
      const int rl = r + r_add;
      P[rl * 32 + (lane & 15)] = (_Float16)p0;
      P[rl * 32 + 16 + (lane & 15)] = (_Float16)p1;
    }
    __builtin_amdgcn_wave_barrier();  // pin DS store -> DS load order

    // Re-load P as an A-fragment (16x32, ld=32) and run PV.
    v16h pa = load_frag(P, 32, lane);
#pragma unroll
    for (int n = 0; n < 4; ++n) acc[n] = wmma_acc(pa, vf[n], acc[n]);

    // Wave 0 drains the DMA; barrier publishes the next buffer to all waves
    // and guarantees everyone is done reading the current one.
    if (w == 0 && have_next) __builtin_amdgcn_s_wait_tensorcnt(0);
    __syncthreads();
  }

  // Normalize and store attention output as f16 (b,t,d) for the final GEMM.
#pragma unroll
  for (int r = 0; r < 8; ++r) {
    const int tok = qbase + r + r_add;
    const float inv = 1.0f / rowsum[r];
#pragma unroll
    for (int n = 0; n < 4; ++n) {
      oh[((size_t)b * SEQ_T + tok) * D_MODEL + h * HEAD_DIM + n * 16 +
         (lane & 15)] = (_Float16)(acc[n][r] * inv);
    }
  }
}

// ---------------------------------------------------------------------------
// Output projection: out = attn_out @ Wo^T + bo   (M=4096, N=768, K=768), f32 out
// ---------------------------------------------------------------------------
__global__ __launch_bounds__(256) void out_proj_kernel(
    const _Float16* __restrict__ oh, const _Float16* __restrict__ woh,
    const float* __restrict__ bo, float* __restrict__ out) {
  const int lane = threadIdx.x & 31;
  const int wid = blockIdx.x * (blockDim.x >> 5) + (threadIdx.x >> 5);
  const int mtile = wid / 12;
  const int nstrip = wid % 12;

  v8f zero8 = {0.f, 0.f, 0.f, 0.f, 0.f, 0.f, 0.f, 0.f};
  v8f acc[4] = {zero8, zero8, zero8, zero8};

  const _Float16* abase = oh + (size_t)(mtile * 16) * D_MODEL;
  for (int kk = 0; kk < D_MODEL; kk += 64) {
    if (kk + 64 < D_MODEL) {
      __builtin_prefetch(abase + kk + 64, 0, 3);
    }
    v16h a0 = load_frag(abase + kk, D_MODEL, lane);
    v16h a1 = load_frag(abase + kk + 32, D_MODEL, lane);
    v16h b0[4], b1[4];
#pragma unroll
    for (int nt = 0; nt < 4; ++nt) {
      const _Float16* bb =
          woh + (size_t)(nstrip * 64 + nt * 16) * D_MODEL + kk;
      b0[nt] = load_frag(bb, D_MODEL, lane);
      b1[nt] = load_frag(bb + 32, D_MODEL, lane);
    }
#pragma unroll
    for (int nt = 0; nt < 4; ++nt) acc[nt] = wmma_acc(a0, b0[nt], acc[nt]);
#pragma unroll
    for (int nt = 0; nt < 4; ++nt) acc[nt] = wmma_acc(a1, b1[nt], acc[nt]);
  }

  const int r_add = (lane & 16) ? 8 : 0;
#pragma unroll
  for (int nt = 0; nt < 4; ++nt) {
    const int col = nstrip * 64 + nt * 16 + (lane & 15);
    const float bval = bo[col];
#pragma unroll
    for (int r = 0; r < 8; ++r) {
      const int tok = mtile * 16 + r + r_add;
      out[(size_t)tok * D_MODEL + col] = acc[nt][r] + bval;
    }
  }
}

// ---------------------------------------------------------------------------
extern "C" void kernel_launch(void* const* d_in, const int* in_sizes, int n_in,
                              void* d_out, int out_size, void* d_ws,
                              size_t ws_size, hipStream_t stream) {
  (void)in_sizes; (void)n_in; (void)out_size; (void)ws_size;
  const float* x = (const float*)d_in[0];
  const int* rel_ids = (const int*)d_in[1];
  const unsigned char* kpm = (const unsigned char*)d_in[2];
  const float* Wq = (const float*)d_in[3];
  const float* bq = (const float*)d_in[4];
  const float* Wk = (const float*)d_in[5];
  const float* bk = (const float*)d_in[6];
  const float* Wv = (const float*)d_in[7];
  const float* bv = (const float*)d_in[8];
  const float* Wo = (const float*)d_in[9];
  const float* bo = (const float*)d_in[10];
  const float* rel_emb = (const float*)d_in[11];
  float* out = (float*)d_out;

  // Workspace layout (f16), ~36.2 MB total
  char* p = (char*)d_ws;
  _Float16* xh = (_Float16*)p;  p += (size_t)NTOK * D_MODEL * 2;
  _Float16* wqh = (_Float16*)p; p += (size_t)D_MODEL * D_MODEL * 2;
  _Float16* wkh = (_Float16*)p; p += (size_t)D_MODEL * D_MODEL * 2;
  _Float16* wvh = (_Float16*)p; p += (size_t)D_MODEL * D_MODEL * 2;
  _Float16* woh = (_Float16*)p; p += (size_t)D_MODEL * D_MODEL * 2;
  _Float16* qh = (_Float16*)p;  p += (size_t)NTOK * D_MODEL * 2;
  _Float16* kh = (_Float16*)p;  p += (size_t)NTOK * D_MODEL * 2;
  _Float16* vT = (_Float16*)p;  p += (size_t)NTOK * D_MODEL * 2;
  _Float16* oh = (_Float16*)p;  p += (size_t)NTOK * D_MODEL * 2;

  const int nx = NTOK * D_MODEL;          // 3145728
  const int nw = D_MODEL * D_MODEL;       // 589824
  cvt_f32_to_f16<<<(nx + 255) / 256, 256, 0, stream>>>(x, xh, nx);
  cvt_f32_to_f16<<<(nw + 255) / 256, 256, 0, stream>>>(Wq, wqh, nw);
  cvt_f32_to_f16<<<(nw + 255) / 256, 256, 0, stream>>>(Wk, wkh, nw);
  cvt_f32_to_f16<<<(nw + 255) / 256, 256, 0, stream>>>(Wv, wvh, nw);
  cvt_f32_to_f16<<<(nw + 255) / 256, 256, 0, stream>>>(Wo, woh, nw);

  // 3 mats * 256 mtiles * 12 nstrips = 9216 waves / 8 per block
  qkv_proj_kernel<<<1152, 256, 0, stream>>>(xh, wqh, wkh, wvh, bq, bk, bv,
                                            qh, kh, vT);
  // 48 (b,h) * 8 q-blocks
  attn_kernel<<<384, 256, 0, stream>>>(qh, kh, vT, rel_ids, kpm, rel_emb, oh);
  // 256 mtiles * 12 nstrips = 3072 waves / 8 per block
  out_proj_kernel<<<384, 256, 0, stream>>>(oh, woh, bo, out);
}
